// AttentionBlock_45509473468664
// MI455X (gfx1250) — compile-verified, matrix operand order checked by reading, not run on hardware
//
#include <hip/hip_runtime.h>
#include <hip/hip_bf16.h>
#include <stdint.h>

typedef unsigned short u16;
typedef __attribute__((ext_vector_type(16))) __bf16 v16bf;
typedef __attribute__((ext_vector_type(8)))  float  v8f;

// ---------- helpers ----------
struct FragBF16 { unsigned u[8]; }; // 16 bf16 packed as 8 dwords

__device__ inline v16bf frag_cast(FragBF16 f) {
    union { FragBF16 s; v16bf v; } c; c.s = f; return c.v;
}

__device__ inline v8f wmma_bf16(FragBF16 a, FragBF16 b, v8f c) {
    return __builtin_amdgcn_wmma_f32_16x16x32_bf16(
        false, frag_cast(a), false, frag_cast(b), (short)0, c, false, false);
}

__device__ inline u16 f2bf(float f) {
    unsigned u = __float_as_uint(f);
    unsigned r = u + 0x7FFFu + ((u >> 16) & 1u);   // round-to-nearest-even
    return (u16)(r >> 16);
}
__device__ inline unsigned pack2(float lo, float hi) {
    return (unsigned)f2bf(lo) | ((unsigned)f2bf(hi) << 16);
}

// Load a 16x32 bf16 A/B fragment (ISA 7.12.2 16-bit layout):
// lane&15 selects the row; lanes 16-31 take K+8; each lane: K [k..k+7] and [k+16..k+23].
__device__ inline FragBF16 load_frag(const u16* __restrict__ p) {
    FragBF16 f;
    uint4 a = *(const uint4*)p;
    uint4 b = *(const uint4*)(p + 16);
    f.u[0]=a.x; f.u[1]=a.y; f.u[2]=a.z; f.u[3]=a.w;
    f.u[4]=b.x; f.u[5]=b.y; f.u[6]=b.z; f.u[7]=b.w;
    return f;
}

// Flat pointers into LDS carry the AS3 byte offset in their low 32 bits.
__device__ inline unsigned lds_off(const void* p) {
    return (unsigned)(uintptr_t)p;
}

// CDNA5 async global->LDS copy (ASYNCcnt-tracked, no VGPR round trip).
__device__ inline void async_b128(unsigned lds_byte_off, const u16* g) {
    asm volatile("global_load_async_to_lds_b128 %0, %1, off"
                 :: "v"(lds_byte_off), "v"(g) : "memory");
}
__device__ inline void wait_async0() {
    asm volatile("s_wait_asynccnt 0" ::: "memory");
}

// ---------- constants ----------
#define CDIM   256
#define NPOS   4096      // 64*64 per batch
#define MTOT   32768     // 8 * 4096
#define SCALE  0.0625f   // 1/sqrt(256)

// ---------- 1) weight transpose + bf16 convert ----------
__global__ void prep_w_kernel(const float* __restrict__ w0, const float* __restrict__ w1,
                              const float* __restrict__ w2, const float* __restrict__ w3,
                              u16* __restrict__ o0, u16* __restrict__ o1,
                              u16* __restrict__ o2, u16* __restrict__ o3) {
    int n = blockIdx.x;     // output row  = input column
    int k = threadIdx.x;    // output col  = input row
    o0[n * CDIM + k] = f2bf(w0[k * CDIM + n]);
    o1[n * CDIM + k] = f2bf(w1[k * CDIM + n]);
    o2[n * CDIM + k] = f2bf(w2[k * CDIM + n]);
    o3[n * CDIM + k] = f2bf(w3[k * CDIM + n]);
}

// ---------- 2) LayerNorm (one wave32 per 256-elem row) ----------
__global__ __launch_bounds__(256) void ln_kernel(const float* __restrict__ x,
                                                 const float* __restrict__ gamma,
                                                 const float* __restrict__ beta,
                                                 u16* __restrict__ h) {
    int row  = blockIdx.x * 8 + (threadIdx.x >> 5);
    int lane = threadIdx.x & 31;
    const float4* xr = (const float4*)(x + (size_t)row * CDIM);
    float4 a = xr[lane];
    float4 b = xr[lane + 32];

    float s  = a.x + a.y + a.z + a.w + b.x + b.y + b.z + b.w;
    float s2 = a.x*a.x + a.y*a.y + a.z*a.z + a.w*a.w
             + b.x*b.x + b.y*b.y + b.z*b.z + b.w*b.w;
    #pragma unroll
    for (int m = 1; m < 32; m <<= 1) {
        s  += __shfl_xor(s,  m, 32);
        s2 += __shfl_xor(s2, m, 32);
    }
    float mean = s * (1.0f / CDIM);
    float var  = s2 * (1.0f / CDIM) - mean * mean;
    float rstd = rsqrtf(var + 1e-3f);

    float4 g0 = ((const float4*)gamma)[lane];
    float4 g1 = ((const float4*)gamma)[lane + 32];
    float4 b0 = ((const float4*)beta)[lane];
    float4 b1 = ((const float4*)beta)[lane + 32];

    float y0 = (a.x - mean) * rstd * g0.x + b0.x;
    float y1 = (a.y - mean) * rstd * g0.y + b0.y;
    float y2 = (a.z - mean) * rstd * g0.z + b0.z;
    float y3 = (a.w - mean) * rstd * g0.w + b0.w;
    float y4 = (b.x - mean) * rstd * g1.x + b1.x;
    float y5 = (b.y - mean) * rstd * g1.y + b1.y;
    float y6 = (b.z - mean) * rstd * g1.z + b1.z;
    float y7 = (b.w - mean) * rstd * g1.w + b1.w;

    unsigned* hr = (unsigned*)(h + (size_t)row * CDIM);
    hr[lane * 2]            = pack2(y0, y1);
    hr[lane * 2 + 1]        = pack2(y2, y3);
    hr[(lane + 32) * 2]     = pack2(y4, y5);
    hr[(lane + 32) * 2 + 1] = pack2(y6, y7);
}

// ---------- 3) WMMA GEMM: D[M,256] = A[M,256] @ Bt[256,256]^T (+bias)  ----------
// MODE 0: store bf16 row-major     MODE 1: store bf16 transposed [256][MTOT]
// MODE 2: store f32 row-major with residual add (final projection)
template<int MODE>
__global__ __launch_bounds__(256) void gemm_kernel(const u16* __restrict__ A,
                                                   const u16* __restrict__ Bt,
                                                   const float* __restrict__ bias,
                                                   void* __restrict__ outp,
                                                   const float* __restrict__ resid) {
    int wave = threadIdx.x >> 5, lane = threadIdx.x & 31;
    int half = lane >> 4,        l15  = lane & 15;
    int wm = wave >> 1, wn = wave & 1;
    int row0 = blockIdx.x * 256 + wm * 64;
    int col0 = blockIdx.y * 64  + wn * 32;

    v8f acc[4][2];
    #pragma unroll
    for (int mt = 0; mt < 4; ++mt)
        #pragma unroll
        for (int nt = 0; nt < 2; ++nt) acc[mt][nt] = {};

    for (int k0 = 0; k0 < CDIM; k0 += 32) {
        FragBF16 af[4], bf[2];
        #pragma unroll
        for (int mt = 0; mt < 4; ++mt)
            af[mt] = load_frag(A + (size_t)(row0 + mt*16 + l15) * CDIM + k0 + half*8);
        #pragma unroll
        for (int nt = 0; nt < 2; ++nt)
            bf[nt] = load_frag(Bt + (size_t)(col0 + nt*16 + l15) * CDIM + k0 + half*8);
        #pragma unroll
        for (int mt = 0; mt < 4; ++mt)
            #pragma unroll
            for (int nt = 0; nt < 2; ++nt)
                acc[mt][nt] = wmma_bf16(af[mt], bf[nt], acc[mt][nt]);
    }

    #pragma unroll
    for (int nt = 0; nt < 2; ++nt) {
        int col = col0 + nt*16 + l15;
        float bv = bias[col];
        #pragma unroll
        for (int mt = 0; mt < 4; ++mt) {
            int rowbase = row0 + mt*16 + half*8;   // D layout: lanes 16-31 hold M+8
            if (MODE == 0) {
                u16* out = (u16*)outp;
                #pragma unroll
                for (int r = 0; r < 8; ++r)
                    out[(size_t)(rowbase + r) * CDIM + col] = f2bf(acc[mt][nt][r] + bv);
            } else if (MODE == 1) {
                u16* out = (u16*)outp;   // [col][MTOT]: 8 consecutive rows per lane
                uint4 pk;
                pk.x = pack2(acc[mt][nt][0] + bv, acc[mt][nt][1] + bv);
                pk.y = pack2(acc[mt][nt][2] + bv, acc[mt][nt][3] + bv);
                pk.z = pack2(acc[mt][nt][4] + bv, acc[mt][nt][5] + bv);
                pk.w = pack2(acc[mt][nt][6] + bv, acc[mt][nt][7] + bv);
                *(uint4*)(out + (size_t)col * MTOT + rowbase) = pk;
            } else {
                float* out = (float*)outp;
                #pragma unroll
                for (int r = 0; r < 8; ++r) {
                    size_t idx = (size_t)(rowbase + r) * CDIM + col;
                    out[idx] = acc[mt][nt][r] + bv + resid[idx];
                }
            }
        }
    }
}

// ---------- 4) Flash attention ----------
// Block = 4 waves = 64 query rows; K/V 32-key tiles staged in LDS with
// double-buffered async global->LDS copies, shared by all 4 waves.
__global__ __launch_bounds__(128) void attn_kernel(const u16* __restrict__ q,
                                                   const u16* __restrict__ k,
                                                   const u16* __restrict__ vT,
                                                   u16* __restrict__ o) {
    __shared__ u16 Kl[2][32][256];   // 2 x 16 KB: K rows j0..j0+31, all channels
    __shared__ u16 Vl[2][256][32];   // 2 x 16 KB: V^T rows (channel), 32 keys
    __shared__ u16 Pl[4][16][32];    // per-wave P bounce (D->A layout)

    int tid  = threadIdx.x;
    int wave = tid >> 5, lane = tid & 31;
    int half = lane >> 4, l15 = lane & 15;
    int b  = blockIdx.x >> 6;
    int m0 = (blockIdx.x & 63) * 64 + wave * 16;

    const u16* qb = q  + ((size_t)b * NPOS + m0) * CDIM;
    const u16* kb = k  + (size_t)b * NPOS * CDIM;
    const u16* vb = vT + (size_t)b * NPOS;           // [c][j] at vb[c*MTOT + j]

    unsigned kbase[2] = { lds_off(&Kl[0][0][0]), lds_off(&Kl[1][0][0]) };
    unsigned vbase[2] = { lds_off(&Vl[0][0][0]), lds_off(&Vl[1][0][0]) };

    // Stage one 32-key block into LDS buffer `buf` (whole workgroup cooperates):
    // K tile is 16 KB contiguous; V^T tile linearizes to 16 KB of b128 chunks.
    auto stage = [&](int buf, int j0) {
        const u16* kg = kb + (size_t)j0 * CDIM;
        #pragma unroll
        for (int i = 0; i < 8; ++i) {
            int id = tid + i * 128;                       // 0..1023
            async_b128(kbase[buf] + id * 16, kg + id * 8);
        }
        #pragma unroll
        for (int i = 0; i < 8; ++i) {
            int id = tid + i * 128;                       // chunk: c = id>>2, q8 = id&3
            async_b128(vbase[buf] + id * 16,
                       vb + (size_t)(id >> 2) * MTOT + j0 + (id & 3) * 8);
        }
    };

    // Resident Q fragments (16 rows x 256 channels)
    FragBF16 qf[8];
    #pragma unroll
    for (int kk = 0; kk < 8; ++kk)
        qf[kk] = load_frag(qb + (size_t)l15 * CDIM + kk*32 + half*8);

    v8f oacc[16];
    #pragma unroll
    for (int t = 0; t < 16; ++t) oacc[t] = {};
    float mrun[8], lrun[8];
    #pragma unroll
    for (int r = 0; r < 8; ++r) { mrun[r] = -3.0e38f; lrun[r] = 0.0f; }

    stage(0, 0);

    for (int jb = 0; jb < NPOS / 32; ++jb) {
        int cur = jb & 1;
        wait_async0();          // this wave's async fills are in LDS
        __syncthreads();        // all waves' fills visible
        if (jb + 1 < NPOS / 32) stage(1 - cur, (jb + 1) * 32);

        // ---- S = Q K^T over the staged 16x32 key block ----
        v8f s0 = {}, s1 = {};
        #pragma unroll
        for (int kk = 0; kk < 8; ++kk) {
            FragBF16 kf0 = load_frag(&Kl[cur][l15     ][kk*32 + half*8]);
            FragBF16 kf1 = load_frag(&Kl[cur][16 + l15][kk*32 + half*8]);
            s0 = wmma_bf16(qf[kk], kf0, s0);
            s1 = wmma_bf16(qf[kk], kf1, s1);
        }
        // ---- online softmax; score row m spans 16 lanes of a half-wave ----
        float corr[8];
        #pragma unroll
        for (int r = 0; r < 8; ++r) {
            float a = s0[r] * SCALE, c = s1[r] * SCALE;
            float t = fmaxf(a, c);
            t = fmaxf(t, __shfl_xor(t, 1, 32));
            t = fmaxf(t, __shfl_xor(t, 2, 32));
            t = fmaxf(t, __shfl_xor(t, 4, 32));
            t = fmaxf(t, __shfl_xor(t, 8, 32));
            float mn = fmaxf(mrun[r], t);
            corr[r] = __expf(mrun[r] - mn);
            float p0 = __expf(a - mn), p1 = __expf(c - mn);
            s0[r] = p0; s1[r] = p1;
            float rs = p0 + p1;
            rs += __shfl_xor(rs, 1, 32);
            rs += __shfl_xor(rs, 2, 32);
            rs += __shfl_xor(rs, 4, 32);
            rs += __shfl_xor(rs, 8, 32);
            lrun[r] = lrun[r] * corr[r] + rs;
            mrun[r] = mn;
        }
        // ---- P tile: D layout -> LDS -> A layout (bf16) ----
        #pragma unroll
        for (int r = 0; r < 8; ++r) {
            Pl[wave][half*8 + r][l15]      = f2bf(s0[r]);
            Pl[wave][half*8 + r][l15 + 16] = f2bf(s1[r]);
        }
        asm volatile("s_wait_dscnt 0" ::: "memory");
        FragBF16 pf = load_frag(&Pl[wave][l15][half*8]);

        // ---- O = O*corr + P @ V  (16 column tiles of 16) ----
        #pragma unroll
        for (int t = 0; t < 16; ++t) {
            FragBF16 vf = load_frag(&Vl[cur][t*16 + l15][half*8]);
            #pragma unroll
            for (int r = 0; r < 8; ++r) oacc[t][r] *= corr[r];
            oacc[t] = wmma_bf16(pf, vf, oacc[t]);
        }
        __syncthreads();        // retire reads of buffer `cur` before refill
    }

    // ---- epilogue: O / l, store bf16 row-major ----
    float inv[8];
    #pragma unroll
    for (int r = 0; r < 8; ++r) inv[r] = 1.0f / lrun[r];
    u16* ob = o + ((size_t)b * NPOS + m0) * CDIM;
    #pragma unroll
    for (int t = 0; t < 16; ++t)
        #pragma unroll
        for (int r = 0; r < 8; ++r)
            ob[(size_t)(half*8 + r) * CDIM + t*16 + l15] = f2bf(oacc[t][r] * inv[r]);
}

// ---------- launcher ----------
extern "C" void kernel_launch(void* const* d_in, const int* in_sizes, int n_in,
                              void* d_out, int out_size, void* d_ws, size_t ws_size,
                              hipStream_t stream) {
    const float* x     = (const float*)d_in[0];
    const float* gamma = (const float*)d_in[1];
    const float* beta  = (const float*)d_in[2];
    const float* wq    = (const float*)d_in[3];
    const float* bq    = (const float*)d_in[4];
    const float* wk    = (const float*)d_in[5];
    const float* bk    = (const float*)d_in[6];
    const float* wv    = (const float*)d_in[7];
    const float* bv    = (const float*)d_in[8];
    const float* wo    = (const float*)d_in[9];
    const float* bo    = (const float*)d_in[10];
    float* out = (float*)d_out;

    char* ws = (char*)d_ws;
    const size_t SZ = (size_t)MTOT * CDIM * sizeof(u16);   // 16.78 MB
    u16* h   = (u16*)(ws + 0 * SZ);
    u16* qb  = (u16*)(ws + 1 * SZ);
    u16* kb  = (u16*)(ws + 2 * SZ);
    u16* vT  = (u16*)(ws + 3 * SZ);
    u16* ob  = (u16*)(ws + 4 * SZ);
    u16* wqT = (u16*)(ws + 5 * SZ);
    u16* wkT = wqT + CDIM * CDIM;
    u16* wvT = wkT + CDIM * CDIM;
    u16* woT = wvT + CDIM * CDIM;

    prep_w_kernel<<<CDIM, CDIM, 0, stream>>>(wq, wk, wv, wo, wqT, wkT, wvT, woT);
    ln_kernel<<<MTOT / 8, 256, 0, stream>>>(x, gamma, beta, h);

    dim3 ggrid(MTOT / 256, CDIM / 64);
    gemm_kernel<0><<<ggrid, 256, 0, stream>>>(h, wqT, bq, (void*)qb, nullptr);
    gemm_kernel<0><<<ggrid, 256, 0, stream>>>(h, wkT, bk, (void*)kb, nullptr);
    gemm_kernel<1><<<ggrid, 256, 0, stream>>>(h, wvT, bv, (void*)vT, nullptr);

    attn_kernel<<<8 * 64, 128, 0, stream>>>(qb, kb, vT, ob);

    gemm_kernel<2><<<ggrid, 256, 0, stream>>>(ob, woT, bo, (void*)out, x);
}
